// EncoderSelfAttention_4896262717992
// MI455X (gfx1250) — matrix operand order repaired and checked by reading, not compiled, not run on hardware
//
#include <hip/hip_runtime.h>

typedef __attribute__((ext_vector_type(16))) _Float16 v16h;
typedef __attribute__((ext_vector_type(8)))  float    v8f;

#define D_MODEL   768
#define N_BATCH   8
#define SEQ_L     1024
#define NH        12
#define HD        64
#define QKV_COLS  2304
#define M_ROWS    8192
#define EPS       1e-6f

// ---------------------------------------------------------------- fragment I/O
union F16x16 { v16h v; _Float16 h[16]; };

// A-matrix 16x32 (MxK) f16 fragment, per CDNA5 ISA layout:
// lanes 0-15: M=lane, K = {0..7, 16..23}; lanes 16-31: M=lane-16, K = {8..15, 24..31}
__device__ inline v16h load_a_frag(const _Float16* __restrict__ p, int stride, int lane) {
  F16x16 u;
  const int m  = lane & 15;
  const int kb = (lane & 16) ? 8 : 0;
  const _Float16* row = p + (size_t)m * stride;
#pragma unroll
  for (int j = 0; j < 4; ++j) {
    u.h[2 * j]     = row[kb + 2 * j];
    u.h[2 * j + 1] = row[kb + 2 * j + 1];
    u.h[8 + 2 * j]     = row[16 + kb + 2 * j];
    u.h[8 + 2 * j + 1] = row[16 + kb + 2 * j + 1];
  }
  return u.v;
}

// B-matrix 32x16 (KxN) f16 fragment. Source tile is stored N-major (row n holds
// the 32 K-values), so each lane reads two runs of 8 f16:
// lanes 0-15: N=lane, K = {0..7, 16..23}; lanes 16-31: N=lane-16, K = {8..15, 24..31}
__device__ inline v16h load_b_frag(const _Float16* __restrict__ p, int stride, int lane) {
  F16x16 u;
  const int n  = lane & 15;
  const int kb = (lane & 16) ? 8 : 0;
  const _Float16* row = p + (size_t)n * stride + kb;
#pragma unroll
  for (int t = 0; t < 16; ++t) u.h[t] = row[(t < 8) ? t : (8 + t)];
  return u.v;
}

__device__ inline v8f wmma_f16(v16h a, v16h b, v8f c) {
  return __builtin_amdgcn_wmma_f32_16x16x32_f16(false, a, false, b, (short)0, c, false, false);
}

// ---- CDNA5 async global->LDS copy (bypasses VGPRs, tracked by ASYNCcnt) ----
// LDS aperture: low 32 bits of the generic pointer == LDS byte address.
__device__ inline void async_copy_b128(const _Float16* gsrc, _Float16* ldst) {
  unsigned lds_addr = (unsigned)(size_t)ldst;
  asm volatile("global_load_async_to_lds_b128 %0, %1, off"
               :: "v"(lds_addr), "v"(gsrc)
               : "memory");
}
__device__ inline void wait_async_all() {
  asm volatile("s_wait_asynccnt 0x0" ::: "memory");
}

// ---------------------------------------------------------------- weight cvt
__global__ __launch_bounds__(256) void cvt_weights_kernel(
    const float* __restrict__ wqkv, const float* __restrict__ wo,
    _Float16* __restrict__ wqkv_h, _Float16* __restrict__ wo_h) {
  size_t i = (size_t)blockIdx.x * 256 + threadIdx.x;
  if (i < (size_t)QKV_COLS * D_MODEL) wqkv_h[i] = (_Float16)wqkv[i];
  if (i < (size_t)D_MODEL * D_MODEL)  wo_h[i]  = (_Float16)wo[i];
}

// ---------------------------------------------------------------- input RMSNorm
__global__ __launch_bounds__(256) void rmsnorm_in_kernel(
    const float* __restrict__ x, _Float16* __restrict__ xn) {
  __shared__ float red[256];
  const int row = blockIdx.x;
  const int t = threadIdx.x;
  const float* xr = x + (size_t)row * D_MODEL;
  float v0 = xr[t], v1 = xr[t + 256], v2 = xr[t + 512];
  red[t] = v0 * v0 + v1 * v1 + v2 * v2;
  __syncthreads();
#pragma unroll
  for (int s = 128; s > 0; s >>= 1) {
    if (t < s) red[t] += red[t + s];
    __syncthreads();
  }
  const float scale = rsqrtf(red[0] * (1.0f / D_MODEL) + EPS);
  _Float16* o = xn + (size_t)row * D_MODEL;
  o[t]       = (_Float16)(v0 * scale);
  o[t + 256] = (_Float16)(v1 * scale);
  o[t + 512] = (_Float16)(v2 * scale);
}

// ---------------------------------------------------------------- QKV GEMM
// C(8192x2304) = Xn(8192x768) * Wqkv(2304x768)^T ; double-buffered async LDS
// staging; epilogue scatters q,k head-major (N,nh,L,hd) and v transposed.
#define BM 128
#define BN 64
#define SA 40   // LDS f16 stride for 32-wide K tiles (pad to dodge bank conflicts)

__global__ __launch_bounds__(256) void qkv_gemm_kernel(
    const _Float16* __restrict__ A, const _Float16* __restrict__ W,
    _Float16* __restrict__ q_ws, _Float16* __restrict__ k_ws,
    _Float16* __restrict__ vt_ws) {
  __shared__ __align__(16) _Float16 ldsA[2 * BM * SA];
  __shared__ __align__(16) _Float16 ldsB[2 * BN * SA];
  const int tid  = threadIdx.x;
  const int lane = tid & 31;
  const int wave = tid >> 5;
  const int wm = wave >> 1, wn = wave & 1;
  const int bm0 = blockIdx.y * BM;
  const int bn0 = blockIdx.x * BN;

  auto stage = [&](int k0, int buf) {
    _Float16* dA = &ldsA[buf * BM * SA];
    _Float16* dB = &ldsB[buf * BN * SA];
#pragma unroll
    for (int p = 0; p < 2; ++p) {                     // A tile: 128x32
      int idx = tid + p * 256;
      int r = idx >> 2, c = (idx & 3) * 8;
      async_copy_b128(&A[(size_t)(bm0 + r) * D_MODEL + k0 + c], &dA[r * SA + c]);
    }
    {                                                 // W tile: 64x32
      int r = tid >> 2, c = (tid & 3) * 8;
      async_copy_b128(&W[(size_t)(bn0 + r) * D_MODEL + k0 + c], &dB[r * SA + c]);
    }
  };

  v8f acc[2][2];
#pragma unroll
  for (int i = 0; i < 2; ++i)
#pragma unroll
    for (int j = 0; j < 2; ++j) acc[i][j] = (v8f){0,0,0,0,0,0,0,0};

  const int nk = D_MODEL / 32;
  stage(0, 0);
  for (int it = 0; it < nk; ++it) {
    const int buf = it & 1;
    wait_async_all();          // tile `it` resident
    __syncthreads();           // all waves done writing buf / done reading buf^1
    if (it + 1 < nk) stage((it + 1) * 32, buf ^ 1);   // prefetch overlaps WMMAs

    const _Float16* cA = &ldsA[buf * BM * SA];
    const _Float16* cB = &ldsB[buf * BN * SA];
    v16h a0 = load_a_frag(&cA[(wm * 32 + 0)  * SA], SA, lane);
    v16h a1 = load_a_frag(&cA[(wm * 32 + 16) * SA], SA, lane);
    v16h b0 = load_b_frag(&cB[(wn * 32 + 0)  * SA], SA, lane);
    v16h b1 = load_b_frag(&cB[(wn * 32 + 16) * SA], SA, lane);
    acc[0][0] = wmma_f16(a0, b0, acc[0][0]);
    acc[0][1] = wmma_f16(a0, b1, acc[0][1]);
    acc[1][0] = wmma_f16(a1, b0, acc[1][0]);
    acc[1][1] = wmma_f16(a1, b1, acc[1][1]);
  }

#pragma unroll
  for (int fm = 0; fm < 2; ++fm)
#pragma unroll
    for (int fn = 0; fn < 2; ++fn)
#pragma unroll
      for (int r = 0; r < 8; ++r) {
        int m = bm0 + wm * 32 + fm * 16 + r + ((lane & 16) ? 8 : 0);
        int c = bn0 + wn * 32 + fn * 16 + (lane & 15);
        float v = acc[fm][fn][r];
        int nidx = m >> 10, l = m & 1023;
        int part = c / D_MODEL, cc = c % D_MODEL;
        int head = cc >> 6, d = cc & 63;
        size_t hb = (size_t)(nidx * NH + head);
        if (part == 0)      q_ws[(hb * SEQ_L + l) * HD + d]  = (_Float16)v;
        else if (part == 1) k_ws[(hb * SEQ_L + l) * HD + d]  = (_Float16)v;
        else                vt_ws[(hb * HD + d) * SEQ_L + l] = (_Float16)v;
      }
}

// ---------------------------------------------------------------- qk-norm
__global__ __launch_bounds__(256) void qknorm_kernel(
    _Float16* __restrict__ q_ws, _Float16* __restrict__ k_ws) {
  const int lane = threadIdx.x & 31, wave = threadIdx.x >> 5;
  const size_t nrows = (size_t)N_BATCH * NH * SEQ_L;   // 98304 per tensor
  size_t row = (size_t)blockIdx.x * 8 + wave;
  _Float16* base = (row < nrows) ? (q_ws + row * HD) : (k_ws + (row - nrows) * HD);
  float v0 = (float)base[lane * 2], v1 = (float)base[lane * 2 + 1];
  float ss = v0 * v0 + v1 * v1;
#pragma unroll
  for (int off = 1; off < 32; off <<= 1) ss += __shfl_xor(ss, off, 32);
  float sc = rsqrtf(ss * (1.0f / HD) + EPS);
  base[lane * 2]     = (_Float16)(v0 * sc);
  base[lane * 2 + 1] = (_Float16)(v1 * sc);
}

// ---------------------------------------------------------------- flash attention
#define QT 128
#define KS 32
#define SK 72
#define SV 40
#define SP 36

__global__ __launch_bounds__(256) void attention_kernel(
    const _Float16* __restrict__ q_ws, const _Float16* __restrict__ k_ws,
    const _Float16* __restrict__ vt_ws, _Float16* __restrict__ attn_ws) {
  __shared__ __align__(16) _Float16 ldsK[2 * KS * SK];   // 32 keys x 64 hd
  __shared__ __align__(16) _Float16 ldsV[2 * HD * SV];   // 64 hd   x 32 keys (Vt)
  __shared__ __align__(16) _Float16 ldsP[8 * 16 * SP];   // per-wave P staging

  const int tid = threadIdx.x, lane = tid & 31, wave = tid >> 5;
  const int head = blockIdx.y, nidx = blockIdx.z;
  const size_t hb = (size_t)(nidx * NH + head);
  const _Float16* Q  = q_ws  + hb * SEQ_L * HD;
  const _Float16* K  = k_ws  + hb * SEQ_L * HD;
  const _Float16* Vt = vt_ws + hb * HD * SEQ_L;
  const int q0 = blockIdx.x * QT + wave * 16;

  v16h qa0 = load_a_frag(Q + (size_t)q0 * HD + 0,  HD, lane);
  v16h qa1 = load_a_frag(Q + (size_t)q0 * HD + 32, HD, lane);

  auto stage = [&](int kt, int buf) {
    int r = tid >> 3, c = (tid & 7) * 8;
    async_copy_b128(&K[(size_t)(kt * KS + r) * HD + c],
                    &ldsK[buf * KS * SK + r * SK + c]);
    int d = tid >> 2, c2 = (tid & 3) * 8;
    async_copy_b128(&Vt[(size_t)d * SEQ_L + kt * KS + c2],
                    &ldsV[buf * HD * SV + d * SV + c2]);
  };

  v8f oacc[4];
#pragma unroll
  for (int c = 0; c < 4; ++c) oacc[c] = (v8f){0,0,0,0,0,0,0,0};
  float mrow[8], lrow[8];
#pragma unroll
  for (int r = 0; r < 8; ++r) { mrow[r] = -1e30f; lrow[r] = 0.0f; }

  _Float16* myP = &ldsP[wave * 16 * SP];

  const int nkt = SEQ_L / KS;
  stage(0, 0);
  for (int kt = 0; kt < nkt; ++kt) {
    const int buf = kt & 1;
    wait_async_all();
    __syncthreads();
    if (kt + 1 < nkt) stage(kt + 1, buf ^ 1);   // async prefetch overlaps compute

    const _Float16* cK = &ldsK[buf * KS * SK];
    const _Float16* cV = &ldsV[buf * HD * SV];

    // S = Q * K^T : two 16-key halves, each 2 WMMAs over hd=64
    v8f s0 = (v8f){0,0,0,0,0,0,0,0}, s1 = (v8f){0,0,0,0,0,0,0,0};
    s0 = wmma_f16(qa0, load_b_frag(&cK[0 * SK + 0],  SK, lane), s0);
    s0 = wmma_f16(qa1, load_b_frag(&cK[0 * SK + 32], SK, lane), s0);
    s1 = wmma_f16(qa0, load_b_frag(&cK[16 * SK + 0],  SK, lane), s1);
    s1 = wmma_f16(qa1, load_b_frag(&cK[16 * SK + 32], SK, lane), s1);

    // online softmax per query row (rows live across 16 lanes of a half-wave)
    float alpha[8];
#pragma unroll
    for (int r = 0; r < 8; ++r) {
      float v0 = s0[r] * 0.125f, v1 = s1[r] * 0.125f;   // 1/sqrt(64)
      float mx = fmaxf(v0, v1);
#pragma unroll
      for (int off = 1; off < 16; off <<= 1) mx = fmaxf(mx, __shfl_xor(mx, off, 16));
      float mnew = fmaxf(mrow[r], mx);
      float a  = __expf(mrow[r] - mnew);
      float p0 = __expf(v0 - mnew);
      float p1 = __expf(v1 - mnew);
      float ps = p0 + p1;
#pragma unroll
      for (int off = 1; off < 16; off <<= 1) ps += __shfl_xor(ps, off, 16);
      lrow[r] = lrow[r] * a + ps;
      mrow[r] = mnew;
      alpha[r] = a;
      int prow = r + ((lane & 16) ? 8 : 0);
      myP[prow * SP + (lane & 15)]      = (_Float16)p0;
      myP[prow * SP + 16 + (lane & 15)] = (_Float16)p1;
    }
#pragma unroll
    for (int c = 0; c < 4; ++c)
#pragma unroll
      for (int r = 0; r < 8; ++r) oacc[c][r] *= alpha[r];

    // O += P * V  (P: 16x32 A-frag, V via transposed tile -> contiguous B-frags)
    v16h pa = load_a_frag(myP, SP, lane);
#pragma unroll
    for (int c = 0; c < 4; ++c)
      oacc[c] = wmma_f16(pa, load_b_frag(&cV[(c * 16) * SV], SV, lane), oacc[c]);
  }

  // normalize + write out in (N, L, nh*hd) layout for the O-proj GEMM
#pragma unroll
  for (int r = 0; r < 8; ++r) {
    float inv = 1.0f / lrow[r];
    int row = q0 + r + ((lane & 16) ? 8 : 0);
    size_t gm = (size_t)nidx * SEQ_L + row;
#pragma unroll
    for (int c = 0; c < 4; ++c) {
      int col = head * HD + c * 16 + (lane & 15);
      attn_ws[gm * D_MODEL + col] = (_Float16)(oacc[c][r] * inv);
    }
  }
}

// ---------------------------------------------------------------- O-proj + residual
__global__ __launch_bounds__(256) void oproj_gemm_kernel(
    const _Float16* __restrict__ A, const _Float16* __restrict__ W,
    const float* __restrict__ inp, const float* __restrict__ oscale,
    float* __restrict__ out) {
  __shared__ __align__(16) _Float16 ldsA[2 * BM * SA];
  __shared__ __align__(16) _Float16 ldsB[2 * BN * SA];
  const int tid  = threadIdx.x;
  const int lane = tid & 31;
  const int wave = tid >> 5;
  const int wm = wave >> 1, wn = wave & 1;
  const int bm0 = blockIdx.y * BM;
  const int bn0 = blockIdx.x * BN;

  auto stage = [&](int k0, int buf) {
    _Float16* dA = &ldsA[buf * BM * SA];
    _Float16* dB = &ldsB[buf * BN * SA];
#pragma unroll
    for (int p = 0; p < 2; ++p) {
      int idx = tid + p * 256;
      int r = idx >> 2, c = (idx & 3) * 8;
      async_copy_b128(&A[(size_t)(bm0 + r) * D_MODEL + k0 + c], &dA[r * SA + c]);
    }
    {
      int r = tid >> 2, c = (tid & 3) * 8;
      async_copy_b128(&W[(size_t)(bn0 + r) * D_MODEL + k0 + c], &dB[r * SA + c]);
    }
  };

  v8f acc[2][2];
#pragma unroll
  for (int i = 0; i < 2; ++i)
#pragma unroll
    for (int j = 0; j < 2; ++j) acc[i][j] = (v8f){0,0,0,0,0,0,0,0};

  const int nk = D_MODEL / 32;
  stage(0, 0);
  for (int it = 0; it < nk; ++it) {
    const int buf = it & 1;
    wait_async_all();
    __syncthreads();
    if (it + 1 < nk) stage((it + 1) * 32, buf ^ 1);

    const _Float16* cA = &ldsA[buf * BM * SA];
    const _Float16* cB = &ldsB[buf * BN * SA];
    v16h a0 = load_a_frag(&cA[(wm * 32 + 0)  * SA], SA, lane);
    v16h a1 = load_a_frag(&cA[(wm * 32 + 16) * SA], SA, lane);
    v16h b0 = load_b_frag(&cB[(wn * 32 + 0)  * SA], SA, lane);
    v16h b1 = load_b_frag(&cB[(wn * 32 + 16) * SA], SA, lane);
    acc[0][0] = wmma_f16(a0, b0, acc[0][0]);
    acc[0][1] = wmma_f16(a0, b1, acc[0][1]);
    acc[1][0] = wmma_f16(a1, b0, acc[1][0]);
    acc[1][1] = wmma_f16(a1, b1, acc[1][1]);
  }

#pragma unroll
  for (int fm = 0; fm < 2; ++fm)
#pragma unroll
    for (int fn = 0; fn < 2; ++fn)
#pragma unroll
      for (int r = 0; r < 8; ++r) {
        int m = bm0 + wm * 32 + fm * 16 + r + ((lane & 16) ? 8 : 0);
        int c = bn0 + wn * 32 + fn * 16 + (lane & 15);
        size_t idx = (size_t)m * D_MODEL + c;
        out[idx] = inp[idx] + acc[fm][fn][r] * oscale[c];
      }
}

// ---------------------------------------------------------------- launch
extern "C" void kernel_launch(void* const* d_in, const int* in_sizes, int n_in,
                              void* d_out, int out_size, void* d_ws, size_t ws_size,
                              hipStream_t stream) {
  const float* input = (const float*)d_in[0];   // (8,32,32,768)
  const float* wqkv  = (const float*)d_in[1];   // (3,768,768)
  const float* wo    = (const float*)d_in[2];   // (768,768)
  const float* osc   = (const float*)d_in[3];   // (768,)
  float* out = (float*)d_out;

  char* ws = (char*)d_ws;
  size_t off = 0;
  auto take = [&](size_t bytes) -> char* {
    char* p = ws + off;
    off += (bytes + 255) & ~(size_t)255;
    return p;
  };
  const size_t act_bytes = (size_t)M_ROWS * D_MODEL * sizeof(_Float16);
  _Float16* xn   = (_Float16*)take(act_bytes);
  _Float16* wqh  = (_Float16*)take((size_t)QKV_COLS * D_MODEL * sizeof(_Float16));
  _Float16* woh  = (_Float16*)take((size_t)D_MODEL * D_MODEL * sizeof(_Float16));
  _Float16* qws  = (_Float16*)take(act_bytes);
  _Float16* kws  = (_Float16*)take(act_bytes);
  _Float16* vtws = (_Float16*)take(act_bytes);
  _Float16* aws  = (_Float16*)take(act_bytes);

  cvt_weights_kernel<<<((size_t)QKV_COLS * D_MODEL + 255) / 256, 256, 0, stream>>>(
      wqkv, wo, wqh, woh);
  rmsnorm_in_kernel<<<M_ROWS, 256, 0, stream>>>(input, xn);

  dim3 g1(QKV_COLS / BN, M_ROWS / BM);
  qkv_gemm_kernel<<<g1, 256, 0, stream>>>(xn, wqh, qws, kws, vtws);

  qknorm_kernel<<<(2u * N_BATCH * NH * SEQ_L) / 8, 256, 0, stream>>>(qws, kws);

  dim3 g2(SEQ_L / QT, NH, N_BATCH);
  attention_kernel<<<g2, 256, 0, stream>>>(qws, kws, vtws, aws);

  dim3 g3(D_MODEL / BN, M_ROWS / BM);
  oproj_gemm_kernel<<<g3, 256, 0, stream>>>(aws, woh, input, osc, out);
}
